// WindowAttention_72018011619784
// MI455X (gfx1250) — matrix operand order repaired and checked by reading, not compiled, not run on hardware
//
#include <hip/hip_runtime.h>
#include <hip/hip_bf16.h>
#include <stdint.h>

// ---------------- problem constants (match reference) ----------------
#define N_TOK   49          // tokens per window (7x7)
#define NP      64          // padded to 4 WMMA tiles
#define DIM     128
#define NHEAD   4
#define HD      32          // head dim
#define NW      64          // windows per mask group
#define NBLK    4096        // total windows
#define THREADS 256         // 8 wave32
#define QSCALE  0.17677669529663687f  // 32^-0.5
#define NN      (N_TOK * N_TOK)       // 2401
#define NBT     (169 * NHEAD)         // bias table elements

typedef __attribute__((ext_vector_type(16))) _Float16 v16h;
typedef __attribute__((ext_vector_type(8)))  _Float16 v8h;
typedef __attribute__((ext_vector_type(4)))  _Float16 v4h;
typedef __attribute__((ext_vector_type(8)))  float    v8f;

// ------------- CDNA5 async copy: global -> LDS (ASYNCcnt path) -------------
// the LDS address comes from the vdst VGPR; generic LDS pointers carry the LDS byte
// offset in their low 32 bits (aperture in high 32), so a truncated cast is
// the correct VDST value.  (ISA 10.x GLOBAL_LOAD_ASYNC_TO_LDS_B32, GV mode)
__device__ __forceinline__ void async_g2l_b32(void* lds_ptr, const void* gptr) {
  uint32_t loff = (uint32_t)(uintptr_t)lds_ptr;
  asm volatile("global_load_async_to_lds_b32 %0, %1, off"
               :: "v"(loff), "v"(gptr) : "memory");
}
__device__ __forceinline__ void wait_async0() {
  asm volatile("s_wait_asynccnt 0x0" ::: "memory");
}

// ------------- WMMA fragment helpers (gfx1250 wave32 layouts) -------------
// A: 16x32 f16, row-major source [m][k]. Lane m=lane&15; k-chunks of 8 at
//    k0+8*(lane>>4) and +16 (ISA 7.12.2 16-bit A table).
__device__ __forceinline__ v16h ld_a_frag(const _Float16* p, int ld, int m0,
                                          int k0, int lane) {
  const int m  = m0 + (lane & 15);
  const int kk = k0 + ((lane >> 4) << 3);
  const _Float16* q = p + m * ld + kk;
  v8h lo = *(const v8h*)(q);
  v8h hi = *(const v8h*)(q + 16);
  return __builtin_shufflevector(lo, hi, 0,1,2,3,4,5,6,7,8,9,10,11,12,13,14,15);
}
// B: 32x16 f16 from COLUMN-major storage [n][k] (i.e. B^T row-major = weight
//    rows). Lane n=lane&15; k-chunks of 8 at k0+8*(lane>>4) and +16
//    (same k order as the A fragment).
__device__ __forceinline__ v16h ld_b_frag(const _Float16* p, int ld, int n0,
                                          int k0, int lane) {
  const int n  = n0 + (lane & 15);
  const int kk = k0 + ((lane >> 4) << 3);
  const _Float16* q = p + n * ld + kk;
  v8h lo = *(const v8h*)(q);
  v8h hi = *(const v8h*)(q + 16);
  return __builtin_shufflevector(lo, hi, 0,1,2,3,4,5,6,7,8,9,10,11,12,13,14,15);
}
// D/C 16x16 f32: lane n=lane&15, VGPR r holds M = r + 8*(lane>=16).
__device__ __forceinline__ void st_d_f32(float* p, int ld, int m0, int n0,
                                         v8f d, int lane) {
  const int n  = n0 + (lane & 15);
  const int mb = m0 + ((lane >> 4) << 3);
#pragma unroll
  for (int r = 0; r < 8; ++r) p[(mb + r) * ld + n] = d[r];
}
__device__ __forceinline__ void st_d_f16_bs(_Float16* p, int ld, int m0, int n0,
                                            v8f d, float bias, float scale,
                                            int lane) {
  const int n  = n0 + (lane & 15);
  const int mb = m0 + ((lane >> 4) << 3);
#pragma unroll
  for (int r = 0; r < 8; ++r)
    p[(mb + r) * ld + n] = (_Float16)((d[r] + bias) * scale);
}
// transposed store: result (m,n) -> p[n][m]  (used to build V^T for PV GEMM)
__device__ __forceinline__ void st_d_f16_T(_Float16* p, int ld, int m0, int n0,
                                           v8f d, float bias, int lane) {
  const int n  = n0 + (lane & 15);
  const int mb = m0 + ((lane >> 4) << 3);
#pragma unroll
  for (int r = 0; r < 8; ++r) p[n * ld + (mb + r)] = (_Float16)(d[r] + bias);
}

__device__ __forceinline__ v8f wmma16(v16h a, v16h b, v8f c) {
  return __builtin_amdgcn_wmma_f32_16x16x32_f16(false, a, false, b, (short)0, c,
                                                false, false);
}

// convert-and-stage helper: one float4 -> 4 f16 in LDS
__device__ __forceinline__ void st4h(_Float16* d, float4 v) {
  v4h h = { (_Float16)v.x, (_Float16)v.y, (_Float16)v.z, (_Float16)v.w };
  *(v4h*)d = h;
}

extern "C" __global__ void __launch_bounds__(THREADS)
swin_window_attn_fused(const float* __restrict__ x,
                       const float* __restrict__ attn_mask,
                       const float* __restrict__ qkv_w,
                       const float* __restrict__ qkv_b,
                       const float* __restrict__ proj_w,
                       const float* __restrict__ proj_b,
                       const float* __restrict__ bias_table,
                       const int*   __restrict__ rel_index,
                       float* __restrict__ out) {
  extern __shared__ __align__(16) char smem[];
  _Float16* s_w   = (_Float16*)(smem);            // [128][128] f16  32KB (qkv chunk / proj_w)
  _Float16* s_x   = (_Float16*)(smem + 32768);    // [64][128]  f16  16KB (x tile, reused as out tile)
  _Float16* s_q   = (_Float16*)(smem + 49152);    // [64][128]  f16  16KB (q*scale)
  _Float16* s_k   = (_Float16*)(smem + 65536);    // [64][128]  f16  16KB
  _Float16* s_vt  = (_Float16*)(smem + 81920);    // [128][64]  f16  16KB (V transposed)
  float*    s_s   = (float*)   (smem + 98304);    // [4][64][64] f32 64KB (scores)
  _Float16* s_p   = (_Float16*)(smem + 163840);   // [4][64][64] f16 32KB (probs)
  float*    s_msk = (float*)   (smem + 196608);   // [49*49] f32  9.6KB (async-staged)
  int*      s_ri  = (int*)     (smem + 206336);   // [49*49] i32  9.6KB (async-staged)
  float*    s_bt  = (float*)   (smem + 216064);   // [169*4] f32  2.7KB (async-staged)

  const int tid  = threadIdx.x;
  const int lane = tid & 31;
  const int wv   = tid >> 5;         // 0..7
  const int b    = blockIdx.x;

  // ---- phase -1: kick off async global->LDS staging of softmax tables ----
  // Overlaps with x staging + the entire QKV GEMM; consumed in phase 2b.
  {
    const float* mg = attn_mask + (size_t)(b & (NW - 1)) * NN;
#pragma unroll
    for (int i = tid; i < NN; i += THREADS) {
      async_g2l_b32(s_msk + i, mg + i);
      async_g2l_b32(s_ri + i, rel_index + i);
    }
#pragma unroll
    for (int i = tid; i < NBT; i += THREADS)
      async_g2l_b32(s_bt + i, bias_table + i);
  }

  // ---- phase 0: stage x -> f16 LDS tile (pad rows 49..63 with zeros) ----
  {
    const float4* xg = (const float4*)(x + (size_t)b * N_TOK * DIM);
#pragma unroll
    for (int i = 0; i < (NP * DIM / 4) / THREADS; ++i) {  // 8 iters
      int idx = tid + i * THREADS;
      int row = idx >> 5;            // 32 float4 per row
      int c4  = idx & 31;
      float4 v = (row < N_TOK) ? xg[row * 32 + c4] : make_float4(0.f, 0.f, 0.f, 0.f);
      st4h(s_x + row * DIM + c4 * 4, v);
    }
    // warm proj_w into L2 for phase 4 (global_prefetch_b8)
    __builtin_prefetch(proj_w + tid * 64, 0, 0);
  }

  // ---- phase 1: QKV GEMM, chunked over q/k/v so the weight buffer is 32KB ----
  for (int c = 0; c < 3; ++c) {
    __syncthreads();                               // prior users of s_w done
    const float4* wg = (const float4*)(qkv_w + (size_t)c * DIM * DIM);
#pragma unroll
    for (int i = 0; i < (DIM * DIM / 4) / THREADS; ++i) {  // 16 iters
      int idx = tid + i * THREADS;
      st4h(s_w + idx * 4, wg[idx]);
    }
    __syncthreads();
    // 64x128 output = 4x8 tiles; wave handles 4 tiles
#pragma unroll
    for (int t = 0; t < 4; ++t) {
      int tile = wv * 4 + t;
      int mt = tile >> 3, nt = tile & 7;
      v8f acc = {};
#pragma unroll
      for (int k0 = 0; k0 < DIM; k0 += 32) {
        v16h a  = ld_a_frag(s_x, DIM, mt * 16, k0, lane);
        v16h bf = ld_b_frag(s_w, DIM, nt * 16, k0, lane);
        acc = wmma16(a, bf, acc);
      }
      int   ncol = nt * 16 + (lane & 15);
      float bias = qkv_b[c * DIM + ncol];
      if (c == 0)      st_d_f16_bs(s_q, DIM, mt * 16, nt * 16, acc, bias, QSCALE, lane);
      else if (c == 1) st_d_f16_bs(s_k, DIM, mt * 16, nt * 16, acc, bias, 1.f, lane);
      else             st_d_f16_T (s_vt, NP, mt * 16, nt * 16, acc, bias, lane);
    }
  }
  __syncthreads();

  // ---- phase 2a: scores = (q*scale) @ k^T per head; K=32 -> 1 WMMA/tile ----
  {
    int h = wv >> 1;
    int mbase = (wv & 1) * 2;
#pragma unroll
    for (int mi = 0; mi < 2; ++mi) {
      int mt = mbase + mi;
#pragma unroll
      for (int nt = 0; nt < 4; ++nt) {
        v8f acc = {};
        v16h a  = ld_a_frag(s_q, DIM, mt * 16, h * HD, lane);
        v16h bf = ld_b_frag(s_k, DIM, nt * 16, h * HD, lane);  // k rows = B cols
        acc = wmma16(a, bf, acc);
        st_d_f32(s_s + h * NP * NP, NP, mt * 16, nt * 16, acc, lane);
      }
    }
  }
  wait_async0();                      // mask / rel_index / bias_table in LDS
  __syncthreads();

  // ---- phase 2b: +rel-pos bias, +shift mask, softmax (1 thread per row) ----
  // All operands now LDS-resident: zero global traffic in this phase.
  {
    int h = tid >> 6;                 // 4 heads
    int m = tid & 63;                 // 64 rows
    float* row = s_s + h * NP * NP + m * NP;
    float mx = -1e30f;
    if (m < N_TOK) {
      const int mo = m * N_TOK;
      for (int n = 0; n < N_TOK; ++n) {
        int   ri = s_ri[mo + n];
        float v  = row[n] + s_bt[ri * NHEAD + h] + s_msk[mo + n];
        row[n] = v;
        mx = fmaxf(mx, v);
      }
    } else {
      for (int n = 0; n < N_TOK; ++n) mx = fmaxf(mx, row[n]);
    }
    float sum = 0.f;
    for (int n = 0; n < N_TOK; ++n) {
      float e = __expf(row[n] - mx);
      sum += e;
      row[n] = e;
    }
    float rs = 1.f / sum;
    _Float16* pr = s_p + h * NP * NP + m * NP;
    for (int n = 0; n < N_TOK; ++n) pr[n] = (_Float16)(row[n] * rs);
    for (int n = N_TOK; n < NP; ++n) pr[n] = (_Float16)0.f;  // padded cols -> 0
  }
  __syncthreads();

  // ---- phase 3: out_head = probs @ V  (B from V^T, contiguous in tokens) ----
  {
    _Float16* s_o = s_x;              // reuse x tile
    int h = wv >> 1;
    int mbase = (wv & 1) * 2;
#pragma unroll
    for (int mi = 0; mi < 2; ++mi) {
      int mt = mbase + mi;
#pragma unroll
      for (int nt = 0; nt < 2; ++nt) {
        v8f acc = {};
#pragma unroll
        for (int k0 = 0; k0 < NP; k0 += 32) {
          v16h a  = ld_a_frag(s_p + h * NP * NP, NP, mt * 16, k0, lane);
          v16h bf = ld_b_frag(s_vt, NP, h * HD + nt * 16, k0, lane);
          acc = wmma16(a, bf, acc);
        }
        st_d_f16_bs(s_o, DIM, mt * 16, h * HD + nt * 16, acc, 0.f, 1.f, lane);
      }
    }
  }
  __syncthreads();

  // ---- phase 4: output projection, epilogue straight to HBM ----
  {
    const float4* wg = (const float4*)proj_w;      // [128][128], L2-resident
#pragma unroll
    for (int i = 0; i < (DIM * DIM / 4) / THREADS; ++i) {
      int idx = tid + i * THREADS;
      st4h(s_w + idx * 4, wg[idx]);
    }
  }
  __syncthreads();
  {
    _Float16* s_o = s_x;
    float* og = out + (size_t)b * N_TOK * DIM;
#pragma unroll
    for (int t = 0; t < 4; ++t) {
      int tile = wv * 4 + t;
      int mt = tile >> 3, nt = tile & 7;
      v8f acc = {};
#pragma unroll
      for (int k0 = 0; k0 < DIM; k0 += 32) {
        v16h a  = ld_a_frag(s_o, DIM, mt * 16, k0, lane);
        v16h bf = ld_b_frag(s_w, DIM, nt * 16, k0, lane);
        acc = wmma16(a, bf, acc);
      }
      int   n  = nt * 16 + (lane & 15);
      int   mb = mt * 16 + ((lane >> 4) << 3);
      float pb = proj_b[n];
#pragma unroll
      for (int r = 0; r < 8; ++r) {
        int row = mb + r;
        if (row < N_TOK) og[row * DIM + n] = acc[r] + pb;  // drop pad rows
      }
    }
  }
}

extern "C" void kernel_launch(void* const* d_in, const int* in_sizes, int n_in,
                              void* d_out, int out_size, void* d_ws,
                              size_t ws_size, hipStream_t stream) {
  (void)in_sizes; (void)n_in; (void)d_ws; (void)ws_size; (void)out_size;
  const float* x          = (const float*)d_in[0];
  const float* attn_mask  = (const float*)d_in[1];
  const float* qkv_w      = (const float*)d_in[2];
  const float* qkv_b      = (const float*)d_in[3];
  const float* proj_w     = (const float*)d_in[4];
  const float* proj_b     = (const float*)d_in[5];
  const float* bias_table = (const float*)d_in[6];
  const int*   rel_index  = (const int*)d_in[7];
  float*       outp       = (float*)d_out;

  const size_t lds_bytes = 219136;   // 214KB < 320KB/WGP on CDNA5
  swin_window_attn_fused<<<NBLK, THREADS, lds_bytes, stream>>>(
      x, attn_mask, qkv_w, qkv_b, proj_w, proj_b, bias_table, rel_index, outp);
}
